// BoltzAffinityHeadReplica_89550068121783
// MI455X (gfx1250) — matrix operand order repaired and checked by reading, not compiled, not run on hardware
//
#include <hip/hip_runtime.h>

#define LC 1024
#define CPAIR 128
#define CSINGLE 512
#define BBINS 64
#define HEADS 8
#define DHEAD 64
#define HIDDEN 256

typedef __attribute__((ext_vector_type(16))) __bf16 bf16x16;
typedef __attribute__((ext_vector_type(8)))  float  f32x8;
typedef __attribute__((ext_vector_type(4)))  float  f32x4;

__device__ __forceinline__ int lane_id() { return (int)(threadIdx.x & 31u); }

__device__ __forceinline__ f32x8 wmma_bf16(bf16x16 a, bf16x16 b, f32x8 c) {
  // D = A(16x32 bf16) * B(32x16 bf16) + C(16x16 f32)
  return __builtin_amdgcn_wmma_f32_16x16x32_bf16(false, a, false, b, (short)0, c,
                                                 false, false);
}

// ---- Fragment loaders (CDNA5 wave32 WMMA VGPR layouts, ISA 7.12.2) ----
// A 16x32 bf16: lane L -> M = L&15 ; K chunks {kb..kb+7} and {kb+16..kb+23}, kb=(L>>4)*8
__device__ __forceinline__ bf16x16 ldA_f32(const float* __restrict__ base, int ld) {
  int l = lane_id(); int m = l & 15; int kb = (l >> 4) * 8;
  const float* r = base + (size_t)m * ld + kb;
  bf16x16 o;
#pragma unroll
  for (int h = 0; h < 2; ++h) {
    const float* p = r + h * 16;
#pragma unroll
    for (int i = 0; i < 8; ++i) o[h * 8 + i] = (__bf16)p[i];
  }
  return o;
}

__device__ __forceinline__ bf16x16 ldA_f32_stats(const float* __restrict__ base, int ld,
                                                 float& sum, float& sq) {
  int l = lane_id(); int m = l & 15; int kb = (l >> 4) * 8;
  const float* r = base + (size_t)m * ld + kb;
  bf16x16 o;
#pragma unroll
  for (int h = 0; h < 2; ++h) {
    const float* p = r + h * 16;
#pragma unroll
    for (int i = 0; i < 8; ++i) {
      float v = p[i];
      sum += v; sq += v * v;
      o[h * 8 + i] = (__bf16)v;
    }
  }
  return o;
}

__device__ __forceinline__ bf16x16 ldA_b16(const __bf16* __restrict__ base, int ld) {
  int l = lane_id(); int m = l & 15; int kb = (l >> 4) * 8;
  const __bf16* r = base + (size_t)m * ld + kb;
  bf16x16 o;
#pragma unroll
  for (int h = 0; h < 2; ++h)
#pragma unroll
    for (int i = 0; i < 8; ++i) o[h * 8 + i] = r[h * 16 + i];
  return o;
}

// B 32x16 bf16 from packed [col][K] layout: lane L -> N = L&15 ; K chunks {kb..kb+7} and {kb+16..kb+23}, kb=(L>>4)*8
__device__ __forceinline__ bf16x16 ldB_pack(const __bf16* __restrict__ base, int ldk) {
  int l = lane_id(); int n = l & 15; int kh = (l >> 4) * 8;
  const __bf16* p = base + (size_t)n * ldk + kh;
  bf16x16 o;
#pragma unroll
  for (int i = 0; i < 16; ++i) o[i] = p[(i & 7) + (i >> 3) * 16];
  return o;
}

// ---------------- prep kernels ----------------
__global__ void k_ln_s(const float* __restrict__ s, const float* __restrict__ g,
                       const float* __restrict__ b, __bf16* __restrict__ sn) {
  int w = threadIdx.x >> 5, l = lane_id();
  int q = blockIdx.x * 8 + w;
  const float* row = s + (size_t)q * CSINGLE;
  float v[16]; float sum = 0.f, sq = 0.f;
#pragma unroll
  for (int i = 0; i < 16; ++i) { float x = row[l + 32 * i]; v[i] = x; sum += x; sq += x * x; }
#pragma unroll
  for (int o = 16; o >= 1; o >>= 1) { sum += __shfl_xor(sum, o, 32); sq += __shfl_xor(sq, o, 32); }
  float mean = sum * (1.f / CSINGLE);
  float rs = rsqrtf(sq * (1.f / CSINGLE) - mean * mean + 1e-5f);
  __bf16* orow = sn + (size_t)q * CSINGLE;
#pragma unroll
  for (int i = 0; i < 16; ++i) {
    int c = l + 32 * i;
    orow[c] = (__bf16)((v[i] - mean) * rs * g[c] + b[c]);
  }
}

__global__ void k_packT(const float* __restrict__ src, __bf16* __restrict__ dst, int R, int C) {
  // dst[c*R + r] = bf16(src[r*C + c])
  for (int idx = blockIdx.x * blockDim.x + threadIdx.x; idx < R * C;
       idx += gridDim.x * blockDim.x) {
    int c = idx / R, r = idx - c * R;
    dst[idx] = (__bf16)src[(size_t)r * C + c];
  }
}

__global__ void k_pack_small(const float* __restrict__ g, const float* __restrict__ bz,
                             const float* __restrict__ wzb, const float* __restrict__ wdist,
                             __bf16* __restrict__ gwp, __bf16* __restrict__ dwp,
                             float* __restrict__ sgw, float* __restrict__ cb) {
  int t = threadIdx.x;
  for (int idx = t; idx < 16 * CPAIR; idx += blockDim.x) {
    int n = idx >> 7, c = idx & 127;
    gwp[idx] = (__bf16)((n < HEADS) ? g[c] * wzb[c * HEADS + n] : 0.f);
  }
  for (int idx = t; idx < 16 * BBINS; idx += blockDim.x) {
    int n = idx >> 6, k = idx & 63;
    dwp[idx] = (__bf16)((n < HEADS) ? wdist[k * HEADS + n] : 0.f);
  }
  if (t < HEADS) {
    float a = 0.f;
    for (int c = 0; c < CPAIR; ++c) a += g[c] * wzb[c * HEADS + t];
    sgw[t] = a;
  } else if (t < 2 * HEADS) {
    int h = t - HEADS; float a = 0.f;
    for (int c = 0; c < CPAIR; ++c) a += bz[c] * wzb[c * HEADS + h];
    cb[h] = a;
  }
}

__global__ void k_prot(const unsigned char* __restrict__ pm, float* __restrict__ prot) {
  int q = blockIdx.x, t = threadIdx.x;
  const unsigned char* row = pm + (size_t)q * LC;
  int any = 0;
#pragma unroll
  for (int i = 0; i < 4; ++i) any |= (int)row[t + 256 * i];
  __shared__ int red[8];
  int wany = (__ballot(any != 0) != 0ull) ? 1 : 0;
  if (lane_id() == 0) red[t >> 5] = wany;
  __syncthreads();
  if (t == 0) {
    int a = 0;
    for (int i = 0; i < 8; ++i) a |= red[i];
    prot[q] = a ? 1.f : 0.f;
  }
}

// ---------------- QKV projection (WMMA) ----------------
__global__ void __launch_bounds__(256)
k_qkv(const __bf16* __restrict__ sn,
      const __bf16* __restrict__ wqp, const __bf16* __restrict__ wkp,
      const __bf16* __restrict__ wvp,
      const float* __restrict__ bq, const float* __restrict__ bk,
      const float* __restrict__ bv,
      __bf16* __restrict__ qb, __bf16* __restrict__ kb, __bf16* __restrict__ vt) {
  int w = threadIdx.x >> 5;
  int row0 = blockIdx.x * 16;
  int c0 = (blockIdx.y * 8 + w) * 16;
  int zi = blockIdx.z;
  const __bf16* wp = (zi == 0) ? wqp : (zi == 1) ? wkp : wvp;
  const float* bias = (zi == 0) ? bq : (zi == 1) ? bk : bv;
  f32x8 acc = {0.f,0.f,0.f,0.f,0.f,0.f,0.f,0.f};
#pragma unroll 4
  for (int i0 = 0; i0 < CSINGLE; i0 += 32) {
    bf16x16 a = ldA_b16(sn + (size_t)row0 * CSINGLE + i0, CSINGLE);
    bf16x16 bm = ldB_pack(wp + (size_t)c0 * CSINGLE + i0, CSINGLE);
    acc = wmma_bf16(a, bm, acc);
  }
  int l = lane_id(), n = l & 15, mb = (l >> 4) * 8;
  float bvn = bias[c0 + n];
#pragma unroll
  for (int j = 0; j < 8; ++j) {
    int r = row0 + mb + j;
    float val = acc[j] + bvn;
    if (zi == 0)      qb[(size_t)r * CSINGLE + c0 + n] = (__bf16)val;
    else if (zi == 1) kb[(size_t)r * CSINGLE + c0 + n] = (__bf16)val;
    else              vt[(size_t)(c0 + n) * LC + r] = (__bf16)val;   // V^T [c][l]
  }
}

// ---- fused: z-LN bias (WMMA), dist bias (WMMA), prior, mask, QK^T (WMMA) ----
__global__ void __launch_bounds__(256)
k_bias_scores(const float* __restrict__ z, const float* __restrict__ dist,
              const unsigned char* __restrict__ pm,
              const float* __restrict__ pc, const float* __restrict__ pa,
              const float* __restrict__ pd,
              const __bf16* __restrict__ gwp, const __bf16* __restrict__ dwp,
              const float* __restrict__ sgwA, const float* __restrict__ cbA,
              const __bf16* __restrict__ qbp, const __bf16* __restrict__ kbp,
              float* __restrict__ logits) {
  __shared__ float biasT[16][16][HEADS];  // [q_local][k_local][h]
  __shared__ int   mkS[16][16];
  int w = threadIdx.x >> 5, l = lane_id();
  int q0 = blockIdx.x * 16, k0 = blockIdx.y * 16;
  int n = l & 15;
  float sgw = (n < HEADS) ? sgwA[n] : 0.f;
  float cbv = (n < HEADS) ? cbA[n] : 0.f;

  // Phase 1: each wave computes pair-bias for 2 query rows x 16 keys
#pragma unroll 1
  for (int qi = 0; qi < 2; ++qi) {
    int qq = w * 2 + qi;
    int q = q0 + qq;
    const float* zrow = z + ((size_t)q * LC + k0) * CPAIR;   // 16 rows of 128 f32
    __builtin_prefetch(zrow, 0, 1);
    f32x8 accZ = {0.f,0.f,0.f,0.f,0.f,0.f,0.f,0.f};
    f32x8 accD = {0.f,0.f,0.f,0.f,0.f,0.f,0.f,0.f};
    float sum = 0.f, sq = 0.f;
#pragma unroll
    for (int c0 = 0; c0 < CPAIR; c0 += 32) {
      bf16x16 a = ldA_f32_stats(zrow + c0, CPAIR, sum, sq);
      bf16x16 b = ldB_pack(gwp + c0, CPAIR);
      accZ = wmma_bf16(a, b, accZ);          // Sum_c z*(g*wzb) per (k,h)
    }
    const float* drow = dist + ((size_t)q * LC + k0) * BBINS;
#pragma unroll
    for (int c0 = 0; c0 < BBINS; c0 += 32) {
      bf16x16 a = ldA_f32(drow + c0, BBINS);
      bf16x16 b = ldB_pack(dwp + c0, BBINS);
      accD = wmma_bf16(a, b, accD);          // dist_bins @ w_dist
    }
    // lane L and L^16 together cover all 128 channels of pair (L&15)
    sum += __shfl_xor(sum, 16, 32);
    sq  += __shfl_xor(sq, 16, 32);
    float mean = sum * (1.f / CPAIR);
    float rs = rsqrtf(sq * (1.f / CPAIR) - mean * mean + 1e-5f);
    int m = l & 15;
    size_t pidx = (size_t)q * LC + k0 + m;
    float pb = logf(pc[pidx] + 1e-6f) - 0.25f * logf(pa[pidx] + 1e-6f)
             - 0.1f * logf(pd[pidx] + 1e-6f);
    int msk = pm[pidx] ? 1 : 0;
    if (l < 16) mkS[qq][l] = msk;
#pragma unroll
    for (int j = 0; j < 8; ++j) {
      int srcm = j + ((l >> 4) << 3);        // C layout: lanes>=16 hold M=8..15
      float mn = __shfl(mean, srcm, 32);
      float rr = __shfl(rs,   srcm, 32);
      float pp = __shfl(pb,   srcm, 32);
      if (n < HEADS)
        biasT[qq][srcm][n] = rr * (accZ[j] - mn * sgw) + cbv + accD[j] + pp;
    }
  }
  __syncthreads();

  // Phase 2: wave w == head h computes Q.K^T for the 16x16 tile, adds bias, masks
  int h = w;
  f32x8 S = {0.f,0.f,0.f,0.f,0.f,0.f,0.f,0.f};
#pragma unroll
  for (int d0 = 0; d0 < DHEAD; d0 += 32) {
    bf16x16 a = ldA_b16(qbp + (size_t)q0 * CSINGLE + h * DHEAD + d0, CSINGLE);
    bf16x16 b = ldB_pack(kbp + (size_t)k0 * CSINGLE + h * DHEAD + d0, CSINGLE);
    S = wmma_bf16(a, b, S);
  }
  int mb = (l >> 4) * 8;
#pragma unroll
  for (int j = 0; j < 8; ++j) {
    int ql = mb + j, kl = n;
    float val = mkS[ql][kl] ? (0.125f * S[j] + biasT[ql][kl][h]) : -1e9f;
    logits[((size_t)(q0 + ql) * HEADS + h) * LC + k0 + kl] = val;   // [q][h][k]
  }
}

// ---------------- softmax over keys, one wave per (q,head), all-register ----------------
__global__ void k_softmax(float* __restrict__ logits) {
  int q = blockIdx.x, w = threadIdx.x >> 5, l = lane_id();
  f32x4* row = (f32x4*)(logits + ((size_t)q * HEADS + w) * LC);
  f32x4 v[8];
  float mx = -3.4e38f;
#pragma unroll
  for (int i = 0; i < 8; ++i) {
    v[i] = row[i * 32 + l];
    mx = fmaxf(mx, fmaxf(fmaxf(v[i].x, v[i].y), fmaxf(v[i].z, v[i].w)));
  }
#pragma unroll
  for (int o = 16; o >= 1; o >>= 1) mx = fmaxf(mx, __shfl_xor(mx, o, 32));
  float sum = 0.f;
#pragma unroll
  for (int i = 0; i < 8; ++i) {
    v[i].x = expf(v[i].x - mx); v[i].y = expf(v[i].y - mx);
    v[i].z = expf(v[i].z - mx); v[i].w = expf(v[i].w - mx);
    sum += v[i].x + v[i].y + v[i].z + v[i].w;
  }
#pragma unroll
  for (int o = 16; o >= 1; o >>= 1) sum += __shfl_xor(sum, o, 32);
  float inv = 1.f / sum;
#pragma unroll
  for (int i = 0; i < 8; ++i) row[i * 32 + l] = v[i] * inv;
}

// write attn_w in reference [q][k][h] order (scalar stores: output is offset by 1 float)
__global__ void k_transpose_out(const float* __restrict__ attn, float* __restrict__ dst) {
  int q = blockIdx.x, t = threadIdx.x;
  const float* src = attn + (size_t)q * HEADS * LC;
  float* d = dst + (size_t)q * LC * HEADS;
#pragma unroll 4
  for (int i = 0; i < 32; ++i) {
    int idx = t + 256 * i;
    int h = idx & 7, k = idx >> 3;
    d[idx] = src[(size_t)h * LC + k];
  }
}

// ---------------- y = attn @ V (WMMA), wave per head, full d=64 per wave ----------------
__global__ void __launch_bounds__(256)
k_av(const float* __restrict__ attn, const __bf16* __restrict__ vt,
     __bf16* __restrict__ yatt) {
  int h = threadIdx.x >> 5, l = lane_id();
  int q0 = blockIdx.x * 16;
  f32x8 a0 = {0.f,0.f,0.f,0.f,0.f,0.f,0.f,0.f};
  f32x8 a1 = a0, a2 = a0, a3 = a0;
  const float* ab = attn + ((size_t)q0 * HEADS + h) * LC;
#pragma unroll 2
  for (int kc = 0; kc < LC; kc += 32) {
    bf16x16 a = ldA_f32(ab + kc, HEADS * LC);
    a0 = wmma_bf16(a, ldB_pack(vt + (size_t)(h * DHEAD +  0) * LC + kc, LC), a0);
    a1 = wmma_bf16(a, ldB_pack(vt + (size_t)(h * DHEAD + 16) * LC + kc, LC), a1);
    a2 = wmma_bf16(a, ldB_pack(vt + (size_t)(h * DHEAD + 32) * LC + kc, LC), a2);
    a3 = wmma_bf16(a, ldB_pack(vt + (size_t)(h * DHEAD + 48) * LC + kc, LC), a3);
  }
  int n = l & 15, mb = (l >> 4) * 8;
#pragma unroll
  for (int j = 0; j < 8; ++j) {
    size_t r = (size_t)(q0 + mb + j) * CSINGLE + h * DHEAD;
    yatt[r +      n] = (__bf16)a0[j];
    yatt[r + 16 + n] = (__bf16)a1[j];
    yatt[r + 32 + n] = (__bf16)a2[j];
    yatt[r + 48 + n] = (__bf16)a3[j];
  }
}

// ---------------- output projection (WMMA), stores y^T [c][q] for pooling ----------------
__global__ void __launch_bounds__(256)
k_oproj(const __bf16* __restrict__ yatt, const __bf16* __restrict__ wop,
        const float* __restrict__ bo, float* __restrict__ yoT) {
  int w = threadIdx.x >> 5, l = lane_id();
  int row0 = blockIdx.x * 16;
  int c0 = (blockIdx.y * 8 + w) * 16;
  f32x8 acc = {0.f,0.f,0.f,0.f,0.f,0.f,0.f,0.f};
#pragma unroll 4
  for (int i0 = 0; i0 < CSINGLE; i0 += 32) {
    acc = wmma_bf16(ldA_b16(yatt + (size_t)row0 * CSINGLE + i0, CSINGLE),
                    ldB_pack(wop + (size_t)c0 * CSINGLE + i0, CSINGLE), acc);
  }
  int n = l & 15, mb = (l >> 4) * 8;
  float bv = bo[c0 + n];
#pragma unroll
  for (int j = 0; j < 8; ++j)
    yoT[(size_t)(c0 + n) * LC + row0 + mb + j] = acc[j] + bv;
}

__global__ void k_pool(const float* __restrict__ yoT, const float* __restrict__ prot,
                       float* __restrict__ pooled) {
  int c = blockIdx.x, t = threadIdx.x;
  const float* col = yoT + (size_t)c * LC;
  float s = 0.f;
#pragma unroll
  for (int i = 0; i < 4; ++i) { int q = t + 256 * i; s += col[q] * prot[q]; }
  __shared__ float red[8];
#pragma unroll
  for (int o = 16; o >= 1; o >>= 1) s += __shfl_xor(s, o, 32);
  if (lane_id() == 0) red[t >> 5] = s;
  __syncthreads();
  if (t == 0) {
    float a = 0.f;
    for (int i = 0; i < 8; ++i) a += red[i];
    pooled[c] = a;
  }
}

__global__ void k_mlp(const float* __restrict__ pooled, const float* __restrict__ prot,
                      const float* __restrict__ g, const float* __restrict__ b,
                      const float* __restrict__ w1, const float* __restrict__ b1,
                      const float* __restrict__ w2, const float* __restrict__ b2,
                      float* __restrict__ out) {
  __shared__ float hs[CSINGLE];
  __shared__ float redA[8], redB[8];
  int t = threadIdx.x, l = t & 31, w = t >> 5;
  float c4 = 0.f;
#pragma unroll
  for (int i = 0; i < 4; ++i) c4 += prot[t + 256 * i];
#pragma unroll
  for (int o = 16; o >= 1; o >>= 1) c4 += __shfl_xor(c4, o, 32);
  if (l == 0) redA[w] = c4;
  __syncthreads();
  float cnt = 0.f;
  for (int i = 0; i < 8; ++i) cnt += redA[i];
  float icnt = 1.f / fmaxf(cnt, 1.f);
  __syncthreads();
  float x0 = pooled[t] * icnt, x1 = pooled[t + 256] * icnt;
  float s = x0 + x1, qq = x0 * x0 + x1 * x1;
#pragma unroll
  for (int o = 16; o >= 1; o >>= 1) { s += __shfl_xor(s, o, 32); qq += __shfl_xor(qq, o, 32); }
  if (l == 0) { redA[w] = s; redB[w] = qq; }
  __syncthreads();
  float S = 0.f, Q = 0.f;
  for (int i = 0; i < 8; ++i) { S += redA[i]; Q += redB[i]; }
  float mean = S * (1.f / CSINGLE);
  float rs = rsqrtf(Q * (1.f / CSINGLE) - mean * mean + 1e-5f);
  hs[t]       = (x0 - mean) * rs * g[t] + b[t];
  hs[t + 256] = (x1 - mean) * rs * g[t + 256] + b[t + 256];
  __syncthreads();
  float acc = b1[t];
  for (int i = 0; i < CSINGLE; ++i) acc += hs[i] * w1[i * HIDDEN + t];
  float part = fmaxf(acc, 0.f) * w2[t];
#pragma unroll
  for (int o = 16; o >= 1; o >>= 1) part += __shfl_xor(part, o, 32);
  __syncthreads();
  if (l == 0) redA[w] = part;
  __syncthreads();
  if (t == 0) {
    float a = 0.f;
    for (int i = 0; i < 8; ++i) a += redA[i];
    out[0] = a + b2[0];
  }
}

extern "C" void kernel_launch(void* const* d_in, const int* in_sizes, int n_in,
                              void* d_out, int out_size, void* d_ws, size_t ws_size,
                              hipStream_t stream) {
  (void)in_sizes; (void)n_in; (void)out_size; (void)ws_size;
  const float* z      = (const float*)d_in[0];
  const float* sprox  = (const float*)d_in[1];
  const float* dist   = (const float*)d_in[2];
  const unsigned char* pmask = (const unsigned char*)d_in[3];
  const float* pcon   = (const float*)d_in[4];
  const float* ppae   = (const float*)d_in[5];
  const float* ppde   = (const float*)d_in[6];
  const float* ln_s_g = (const float*)d_in[7];
  const float* ln_s_b = (const float*)d_in[8];
  const float* w_q    = (const float*)d_in[9];
  const float* b_q    = (const float*)d_in[10];
  const float* w_k    = (const float*)d_in[11];
  const float* b_k    = (const float*)d_in[12];
  const float* w_v    = (const float*)d_in[13];
  const float* b_v    = (const float*)d_in[14];
  const float* ln_z_g = (const float*)d_in[15];
  const float* ln_z_b = (const float*)d_in[16];
  const float* w_zb   = (const float*)d_in[17];
  const float* w_dist = (const float*)d_in[18];
  const float* w_o    = (const float*)d_in[19];
  const float* b_o    = (const float*)d_in[20];
  const float* mlp_g  = (const float*)d_in[21];
  const float* mlp_b  = (const float*)d_in[22];
  const float* mlp_w1 = (const float*)d_in[23];
  const float* mlp_b1 = (const float*)d_in[24];
  const float* mlp_w2 = (const float*)d_in[25];
  const float* mlp_b2 = (const float*)d_in[26];
  float* out = (float*)d_out;

  char* W = (char*)d_ws;
  float*  logits = (float*) (W);                         // 32 MB [q][h][k]
  __bf16* sn     = (__bf16*)(W + 33554432);              // 1 MB
  __bf16* qb     = (__bf16*)(W + 34603008);              // 1 MB
  __bf16* kb     = (__bf16*)(W + 35651584);              // 1 MB
  __bf16* vt     = (__bf16*)(W + 36700160);              // 1 MB  V^T [c][l]
  __bf16* yatt   = (__bf16*)(W + 37748736);              // 1 MB
  __bf16* wqp    = (__bf16*)(W + 38797312);              // 512 KB each, packed [out][in]
  __bf16* wkp    = (__bf16*)(W + 39321600);
  __bf16* wvp    = (__bf16*)(W + 39845888);
  __bf16* wop    = (__bf16*)(W + 40370176);
  float*  yoT    = (float*) (W + 40894464);              // 2 MB y^T [c][q]
  __bf16* gwp    = (__bf16*)(W + 42991616);              // 4 KB [16][128]
  __bf16* dwp    = (__bf16*)(W + 42995712);              // 2 KB [16][64]
  float*  sgw    = (float*) (W + 42997760);              // 32 B
  float*  cbb    = (float*) (W + 42997824);              // 32 B
  float*  prot   = (float*) (W + 42997888);              // 4 KB
  float*  pooled = (float*) (W + 43001984);              // 2 KB

  hipLaunchKernelGGL(k_ln_s, dim3(128), dim3(256), 0, stream, sprox, ln_s_g, ln_s_b, sn);
  hipLaunchKernelGGL(k_packT, dim3(256), dim3(256), 0, stream, w_q, wqp, CSINGLE, CSINGLE);
  hipLaunchKernelGGL(k_packT, dim3(256), dim3(256), 0, stream, w_k, wkp, CSINGLE, CSINGLE);
  hipLaunchKernelGGL(k_packT, dim3(256), dim3(256), 0, stream, w_v, wvp, CSINGLE, CSINGLE);
  hipLaunchKernelGGL(k_packT, dim3(256), dim3(256), 0, stream, w_o, wop, CSINGLE, CSINGLE);
  hipLaunchKernelGGL(k_pack_small, dim3(1), dim3(256), 0, stream,
                     ln_z_g, ln_z_b, w_zb, w_dist, gwp, dwp, sgw, cbb);
  hipLaunchKernelGGL(k_prot, dim3(LC), dim3(256), 0, stream, pmask, prot);
  hipLaunchKernelGGL(k_qkv, dim3(64, 4, 3), dim3(256), 0, stream,
                     sn, wqp, wkp, wvp, b_q, b_k, b_v, qb, kb, vt);
  hipLaunchKernelGGL(k_bias_scores, dim3(64, 64), dim3(256), 0, stream,
                     z, dist, pmask, pcon, ppae, ppde, gwp, dwp, sgw, cbb, qb, kb, logits);
  hipLaunchKernelGGL(k_softmax, dim3(LC), dim3(256), 0, stream, logits);
  hipLaunchKernelGGL(k_transpose_out, dim3(LC), dim3(256), 0, stream, logits, out + 1);
  hipLaunchKernelGGL(k_av, dim3(64), dim3(256), 0, stream, logits, vt, yatt);
  hipLaunchKernelGGL(k_oproj, dim3(64, 4), dim3(256), 0, stream, yatt, wop, b_o, yoT);
  hipLaunchKernelGGL(k_pool, dim3(CSINGLE), dim3(256), 0, stream, yoT, prot, pooled);
  hipLaunchKernelGGL(k_mlp, dim3(1), dim3(256), 0, stream,
                     pooled, prot, mlp_g, mlp_b, mlp_w1, mlp_b1, mlp_w2, mlp_b2, out);
}
